// Tacotron2_75265006895475
// MI455X (gfx1250) — matrix operand order repaired and checked by reading, not compiled, not run on hardware
//
#include <hip/hip_runtime.h>
#include <hip/hip_bf16.h>

// ---------------- types ----------------
typedef __bf16 v16bf __attribute__((ext_vector_type(16)));
typedef float  v8f   __attribute__((ext_vector_type(8)));

union Frag {
    v16bf v;
    uint4 u[2];   // 32 bytes = 16 bf16
};

#define WMMA_BF16(a, b, c) \
    __builtin_amdgcn_wmma_f32_16x16x32_bf16(false, (a), false, (b), (short)0, (c), false, false)

static constexpr int B_   = 64;
static constexpr int T_   = 600;
static constexpr int NMEL = 80;
static constexpr int ENC  = 544;   // ENC_POS
static constexpr int PRE  = 256;
static constexpr int RNN  = 1024;
static constexpr int XDIM = 800;   // PRENET + ENC
static constexpr int PROJK= 1568;  // RNN + ENC
static constexpr int NWG  = 32;    // each WG owns 32 hidden columns

// ---------------- helpers ----------------
__device__ __forceinline__ unsigned short f32_to_bf16(float f) {
    unsigned u = __float_as_uint(f);
    u += 0x7FFFu + ((u >> 16) & 1u);     // round-to-nearest-even
    return (unsigned short)(u >> 16);
}

__device__ __forceinline__ unsigned hashu(unsigned x) {
    x ^= x >> 16; x *= 0x7feb352du;
    x ^= x >> 15; x *= 0x846ca68bu;
    x ^= x >> 16; return x;
}

__device__ __forceinline__ float sigm(float x) { return 1.0f / (1.0f + __expf(-x)); }

// A fragment: row-major A[row][k], lda multiple of 8.
// lane L: row = m0 + (L&15); chunks at k0 + ((L>>4)*8) and +16.
__device__ __forceinline__ Frag load_a(const unsigned short* A, int lda, int m0, int k0, int lane) {
    Frag f;
    const uint4* p = (const uint4*)(A + (size_t)(m0 + (lane & 15)) * lda + k0 + ((lane >> 4) << 3));
    f.u[0] = p[0];
    f.u[1] = p[2];
    return f;
}

// B fragment for gates[n] = sum_k x[k]*W[n][k]: B column n == W row n (contiguous).
// lane L: n = n0 + (L&15); chunks at k0 + ((L>>4)*8) and +16.
__device__ __forceinline__ Frag load_b(const unsigned short* W, int ldk, int n0, int k0, int lane) {
    Frag f;
    const uint4* p = (const uint4*)(W + (size_t)(n0 + (lane & 15)) * ldk + k0 + ((lane >> 4) << 3));
    f.u[0] = p[0];
    f.u[1] = p[2];
    return f;
}

// Four M-tiles (full M=64) sharing one B-fragment stream: 4 WMMAs per B load.
// k-loop unrolled x2 so the scheduler can clause-group ~20 b128 loads across 8 WMMAs.
__device__ __forceinline__ void gemm_quad(v8f acc[4],
                                          const unsigned short* __restrict__ A, int lda,
                                          const unsigned short* __restrict__ W, int ldk, int n0,
                                          int kTot, int lane) {
#pragma unroll 2
    for (int k0 = 0; k0 < kTot; k0 += 32) {
        Frag b = load_b(W, ldk, n0, k0, lane);
#pragma unroll
        for (int m = 0; m < 4; ++m) {
            Frag a = load_a(A, lda, m * 16, k0, lane);
            acc[m] = WMMA_BF16(a.v, b.v, acc[m]);
        }
    }
}

// ---------------- init / conversion kernels ----------------
__global__ void init_state(unsigned short* h1buf, unsigned short* h2buf, unsigned* cnt) {
    int i = blockIdx.x * blockDim.x + threadIdx.x;
    if (i == 0) *cnt = 0u;
    for (int j = i; j < 2 * B_ * RNN; j += gridDim.x * blockDim.x) {
        h1buf[j] = 0; h2buf[j] = 0;
    }
}

__global__ void cvt_f32_bf16(const float* __restrict__ src, unsigned short* __restrict__ dst, int n) {
    for (int i = blockIdx.x * blockDim.x + threadIdx.x; i < n; i += gridDim.x * blockDim.x)
        dst[i] = f32_to_bf16(src[i]);
}

// Wp1 [256][80] fp32 -> [256][96] bf16, zero-padded K tail
__global__ void cvt_pad_wp1(const float* __restrict__ src, unsigned short* __restrict__ dst) {
    int n = 256 * 96;
    for (int i = blockIdx.x * blockDim.x + threadIdx.x; i < n; i += gridDim.x * blockDim.x) {
        int r = i / 96, k = i % 96;
        dst[i] = (k < 80) ? f32_to_bf16(src[r * 80 + k]) : (unsigned short)0;
    }
}

// decoder_inputs [64][80][600] -> decbf [600][64][96] bf16 (go-frame at t=0, shifted by 1)
__global__ void build_decbf(const float* __restrict__ dec, unsigned short* __restrict__ out) {
    int n = T_ * B_ * 96;
    for (int i = blockIdx.x * blockDim.x + threadIdx.x; i < n; i += gridDim.x * blockDim.x) {
        int t = i / (B_ * 96);
        int b = (i / 96) % B_;
        int k = i % 96;
        float v = (t == 0 || k >= 80) ? 0.0f : dec[((size_t)b * 80 + k) * T_ + (t - 1)];
        out[i] = f32_to_bf16(v);
    }
}

// duration_outputs [64][600][544] -> xbuf[t][b][256 + j] bf16
__global__ void build_dur(const float* __restrict__ dur, unsigned short* __restrict__ xbuf) {
    int n = T_ * B_ * ENC;
    for (int i = blockIdx.x * blockDim.x + threadIdx.x; i < n; i += gridDim.x * blockDim.x) {
        int t = i / (B_ * ENC);
        int b = (i / ENC) % B_;
        int j = i % ENC;
        xbuf[((size_t)t * B_ + b) * XDIM + PRE + j] =
            f32_to_bf16(dur[((size_t)b * T_ + t) * ENC + j]);
    }
}

// ---------------- prenet: GEMM(38400 x 256, K) + relu + dropout(0.5) ----------------
// grid = 2400 blocks (one 16-row M tile each), 256 threads = 8 waves, 2 N-tiles per wave.
__global__ void __launch_bounds__(256) prenet_gemm(const unsigned short* __restrict__ A, int lda,
                                                   const unsigned short* __restrict__ W, int kTot,
                                                   unsigned short* __restrict__ out, int ldo,
                                                   unsigned seed) {
    const int lane = threadIdx.x & 31, wave = threadIdx.x >> 5;
    const int m0 = blockIdx.x * 16;
    for (int nt = wave * 2; nt < wave * 2 + 2; ++nt) {
        const int n0 = nt * 16;
        v8f acc = {};
        for (int k0 = 0; k0 < kTot; k0 += 32) {
            Frag a = load_a(A, lda, m0, k0, lane);
            Frag b = load_b(W, kTot, n0, k0, lane);
            acc = WMMA_BF16(a.v, b.v, acc);
        }
        const int col = n0 + (lane & 15);
#pragma unroll
        for (int j = 0; j < 8; ++j) {
            int row = m0 + j + ((lane >> 4) << 3);
            float v = acc[j];
            v = v > 0.0f ? v : 0.0f;                                  // relu
            unsigned h = hashu(seed ^ (unsigned)(row * 256 + col));   // dropout p=0.5
            v = (h & 1u) ? v * 2.0f : 0.0f;
            out[(size_t)row * ldo + col] = f32_to_bf16(v);
        }
    }
}

// ---------------- persistent decoder kernel ----------------
__device__ __forceinline__ void gbar(unsigned* cnt, unsigned target) {
    __syncthreads();
    if (threadIdx.x == 0) {
        __threadfence();
        __hip_atomic_fetch_add(cnt, 1u, __ATOMIC_ACQ_REL, __HIP_MEMORY_SCOPE_AGENT);
        while (__hip_atomic_load(cnt, __ATOMIC_ACQUIRE, __HIP_MEMORY_SCOPE_AGENT) < target)
            __builtin_amdgcn_s_sleep(2);
    }
    __syncthreads();
}

// stage gates to LDS: layout [4 gates][64 batch][32 cols] f32 (32 KB)
__device__ __forceinline__ void store_lds32(float* lds, int q, int ct, int m0, v8f acc, int lane) {
    const int col = ct * 16 + (lane & 15);
#pragma unroll
    for (int j = 0; j < 8; ++j) {
        int brow = m0 + j + ((lane >> 4) << 3);
        lds[(q * 64 + brow) * 32 + col] = acc[j];
    }
}

__device__ __forceinline__ void lstm_update32(const float* lds, float* cr,
                                              const float* __restrict__ bih,
                                              const float* __restrict__ bhh,
                                              int n0, unsigned short* __restrict__ hw, int tid) {
#pragma unroll
    for (int i = 0; i < 8; ++i) {
        int idx = tid * 8 + i;           // 0..2047 over (b, col)
        int b = idx >> 5, col = idx & 31;
        int gc = n0 + col;
        float ig = lds[(0 * 64 + b) * 32 + col] + bih[gc]            + bhh[gc];
        float fg = lds[(1 * 64 + b) * 32 + col] + bih[1024 + gc]     + bhh[1024 + gc];
        float gg = lds[(2 * 64 + b) * 32 + col] + bih[2048 + gc]     + bhh[2048 + gc];
        float og = lds[(3 * 64 + b) * 32 + col] + bih[3072 + gc]     + bhh[3072 + gc];
        float c = sigm(fg) * cr[i] + sigm(ig) * tanhf(gg);
        float h = sigm(og) * tanhf(c);
        cr[i] = c;
        hw[(size_t)b * RNN + n0 + col] = f32_to_bf16(h);
    }
}

__global__ void __launch_bounds__(256) decoder_kernel(
    const unsigned short* __restrict__ xbuf,
    const unsigned short* __restrict__ Wih1, const unsigned short* __restrict__ Whh1,
    const unsigned short* __restrict__ Wih2, const unsigned short* __restrict__ Whh2,
    const unsigned short* __restrict__ Wproj,
    const float* __restrict__ bih1, const float* __restrict__ bhh1,
    const float* __restrict__ bih2, const float* __restrict__ bhh2,
    const float* __restrict__ bproj,
    unsigned short* __restrict__ h1buf, unsigned short* __restrict__ h2buf,
    float* __restrict__ out, unsigned* cnt) {

    __shared__ float lds[4 * 64 * 32];   // 4 gates x 64 batch x 32 cols (32 KB)

    const int tid = threadIdx.x, lane = tid & 31, wave = tid >> 5, wg = blockIdx.x;
    const int n0 = wg * 32;              // this WG's hidden-column block (32 cols)
    float c1r[8] = {0.f, 0.f, 0.f, 0.f, 0.f, 0.f, 0.f, 0.f};
    float c2r[8] = {0.f, 0.f, 0.f, 0.f, 0.f, 0.f, 0.f, 0.f};
    unsigned phase = 0;

    const int q  = wave >> 1;            // gate index (i/f/g/o)
    const int ct = wave & 1;             // column-tile within WG's 32 cols
    const int nq = q * RNN + n0 + ct * 16;  // gate column block in [4096]

    for (int t = 0; t < T_; ++t) {
        const unsigned short* h1r = h1buf + (size_t)(t & 1) * (B_ * RNN);
        unsigned short*       h1w = h1buf + (size_t)((t + 1) & 1) * (B_ * RNN);
        const unsigned short* h2r = h2buf + (size_t)(t & 1) * (B_ * RNN);
        unsigned short*       h2w = h2buf + (size_t)((t + 1) & 1) * (B_ * RNN);
        const unsigned short* xt  = xbuf + (size_t)t * B_ * XDIM;

        // ---- Phase A: LSTM1 gates = x @ Wih1^T + h1 @ Whh1^T ----
        {
            v8f acc[4] = {};
            gemm_quad(acc, xt,  XDIM, Wih1, XDIM, nq, XDIM, lane);
            gemm_quad(acc, h1r, RNN,  Whh1, RNN,  nq, RNN,  lane);
#pragma unroll
            for (int m = 0; m < 4; ++m) store_lds32(lds, q, ct, m * 16, acc[m], lane);
            // warm next timestep's x slab into L2 while we have slack
            if (t + 1 < T_)
                __builtin_prefetch(xbuf + (size_t)(t + 1) * B_ * XDIM + tid * 64, 0, 1);
        }
        __syncthreads();
        lstm_update32(lds, c1r, bih1, bhh1, n0, h1w, tid);
        gbar(cnt, ++phase * NWG);

        // ---- Phase B: LSTM2 gates = h1_new @ Wih2^T + h2 @ Whh2^T ----
        {
            v8f acc[4] = {};
            gemm_quad(acc, h1w, RNN, Wih2, RNN, nq, RNN, lane);
            gemm_quad(acc, h2r, RNN, Whh2, RNN, nq, RNN, lane);
#pragma unroll
            for (int m = 0; m < 4; ++m) store_lds32(lds, q, ct, m * 16, acc[m], lane);
        }
        __syncthreads();
        lstm_update32(lds, c2r, bih2, bhh2, n0, h2w, tid);
        gbar(cnt, ++phase * NWG);

        // ---- Phase C: mel = concat(h2_new, dur) @ Wproj^T + b ----
        // WGs 0..2 x 8 waves: mel tile = wg*32 + ct*16 (<80), batch tile = (wave>>1)*16.
        {
            const int melBase = n0 + ct * 16;
            if (wg < 3 && melBase < NMEL) {
                const int m0  = (wave >> 1) * 16;    // batch tile
                const int mel = melBase + (lane & 15);
                v8f acc = {};
                // h2 part: k in [0, 1024)
#pragma unroll 2
                for (int k0 = 0; k0 < RNN; k0 += 32) {
                    Frag b = load_b(Wproj, PROJK, melBase, k0, lane);
                    Frag a = load_a(h2w, RNN, m0, k0, lane);
                    acc = WMMA_BF16(a.v, b.v, acc);
                }
                // dur part: k in [1024, 1568) -> x cols 256..799
#pragma unroll 2
                for (int k0 = RNN; k0 < PROJK; k0 += 32) {
                    Frag b = load_b(Wproj, PROJK, melBase, k0, lane);
                    Frag a = load_a(xt, XDIM, m0, k0 - (RNN - PRE), lane);
                    acc = WMMA_BF16(a.v, b.v, acc);
                }
#pragma unroll
                for (int j = 0; j < 8; ++j) {
                    int b = m0 + j + ((lane >> 4) << 3);
                    out[(size_t)b * (NMEL * T_) + (size_t)mel * T_ + t] = acc[j] + bproj[mel];
                }
            }
        }
        // no barrier needed: next write targets alternate double buffers
    }
}

// ---------------- host launch ----------------
extern "C" void kernel_launch(void* const* d_in, const int* in_sizes, int n_in,
                              void* d_out, int out_size, void* d_ws, size_t ws_size,
                              hipStream_t stream) {
    (void)in_sizes; (void)n_in; (void)out_size; (void)ws_size;

    const float* dur   = (const float*)d_in[0];   // [64][600][544]
    const float* dec   = (const float*)d_in[1];   // [64][80][600]
    const float* Wp1   = (const float*)d_in[3];   // [256][80]
    const float* Wp2   = (const float*)d_in[4];   // [256][256]
    const float* Wih1  = (const float*)d_in[5];   // [4096][800]
    const float* Whh1  = (const float*)d_in[6];   // [4096][1024]
    const float* bih1  = (const float*)d_in[7];
    const float* bhh1  = (const float*)d_in[8];
    const float* Wih2  = (const float*)d_in[9];   // [4096][1024]
    const float* Whh2  = (const float*)d_in[10];  // [4096][1024]
    const float* bih2  = (const float*)d_in[11];
    const float* bhh2  = (const float*)d_in[12];
    const float* Wproj = (const float*)d_in[13];  // [80][1568]
    const float* bproj = (const float*)d_in[14];
    float* out = (float*)d_out;                   // [64][80][600]

    // workspace carve-up (256B aligned)
    unsigned char* base = (unsigned char*)d_ws;
    size_t cur = 0;
    auto alloc = [&](size_t bytes) -> void* {
        void* p = base + cur;
        cur += (bytes + 255) & ~(size_t)255;
        return p;
    };
    unsigned*       cnt    = (unsigned*)alloc(256);
    unsigned short* Wih1b  = (unsigned short*)alloc((size_t)4096 * 800  * 2);
    unsigned short* Whh1b  = (unsigned short*)alloc((size_t)4096 * 1024 * 2);
    unsigned short* Wih2b  = (unsigned short*)alloc((size_t)4096 * 1024 * 2);
    unsigned short* Whh2b  = (unsigned short*)alloc((size_t)4096 * 1024 * 2);
    unsigned short* Wp1b   = (unsigned short*)alloc((size_t)256 * 96 * 2);
    unsigned short* Wp2b   = (unsigned short*)alloc((size_t)256 * 256 * 2);
    unsigned short* Wprojb = (unsigned short*)alloc((size_t)80 * 1568 * 2);
    unsigned short* decbf  = (unsigned short*)alloc((size_t)T_ * B_ * 96 * 2);
    unsigned short* interm = (unsigned short*)alloc((size_t)T_ * B_ * 256 * 2);
    unsigned short* xbuf   = (unsigned short*)alloc((size_t)T_ * B_ * XDIM * 2);
    unsigned short* h1buf  = (unsigned short*)alloc((size_t)2 * B_ * RNN * 2);
    unsigned short* h2buf  = (unsigned short*)alloc((size_t)2 * B_ * RNN * 2);

    init_state<<<64, 256, 0, stream>>>(h1buf, h2buf, cnt);

    cvt_f32_bf16<<<2048, 256, 0, stream>>>(Wih1,  Wih1b,  4096 * 800);
    cvt_f32_bf16<<<2048, 256, 0, stream>>>(Whh1,  Whh1b,  4096 * 1024);
    cvt_f32_bf16<<<2048, 256, 0, stream>>>(Wih2,  Wih2b,  4096 * 1024);
    cvt_f32_bf16<<<2048, 256, 0, stream>>>(Whh2,  Whh2b,  4096 * 1024);
    cvt_f32_bf16<<<256,  256, 0, stream>>>(Wp2,   Wp2b,   256 * 256);
    cvt_f32_bf16<<<512,  256, 0, stream>>>(Wproj, Wprojb, 80 * 1568);
    cvt_pad_wp1<<<96, 256, 0, stream>>>(Wp1, Wp1b);
    build_decbf<<<2048, 256, 0, stream>>>(dec, decbf);
    build_dur<<<4096, 256, 0, stream>>>(dur, xbuf);

    // prenet layer 1: [38400 x 96] @ [96 x 256] -> interm (relu+dropout, bf16)
    prenet_gemm<<<T_ * B_ / 16, 256, 0, stream>>>(decbf, 96, Wp1b, 96, interm, 256, 0x9E3779B9u);
    // prenet layer 2: [38400 x 256] @ [256 x 256] -> xbuf[:, 0:256]
    prenet_gemm<<<T_ * B_ / 16, 256, 0, stream>>>(interm, 256, Wp2b, 256, xbuf, XDIM, 0x85EBCA6Bu);

    // persistent recurrent decoder
    decoder_kernel<<<NWG, 256, 0, stream>>>(xbuf,
                                            Wih1b, Whh1b, Wih2b, Whh2b, Wprojb,
                                            bih1, bhh1, bih2, bhh2, bproj,
                                            h1buf, h2buf, out, cnt);
}
